// DotProductAttention_76338748719461
// MI455X (gfx1250) — matrix operand order repaired and checked by reading, not compiled, not run on hardware
//
#include <hip/hip_runtime.h>
#include <hip/hip_bf16.h>

typedef __attribute__((ext_vector_type(16))) _Float16 v16h;
typedef __attribute__((ext_vector_type(8)))  _Float16 v8h;
typedef __attribute__((ext_vector_type(8)))  float    v8f;
typedef __attribute__((ext_vector_type(4)))  unsigned int v4u;
typedef __attribute__((ext_vector_type(8)))  int      v8i;
typedef __attribute__((ext_vector_type(4)))  int      v4i;

#define B_DIM 16
#define LQ 2048
#define LK 2048
#define DH 64

// ---------------------------------------------------------------------------
// Tensor Data Mover: issue a 2-D (or 1-D with tile_h==1) tile load to LDS.
// Descriptor layout per CDNA5 ISA (D# groups 0/1).
// data_size = 4 bytes (dword units). Tracked by TENSORcnt.
// ---------------------------------------------------------------------------
__device__ __forceinline__ void tdm_load_2d(unsigned int lds_off, const void* gsrc,
                                            unsigned int tile_w_dw, unsigned int tile_h,
                                            unsigned int stride_dw,
                                            unsigned int tensor_w_dw, unsigned int tensor_h) {
    unsigned long long ga = (unsigned long long)(uintptr_t)gsrc;
    v4u g0;
    g0.x = 1u;                                              // count=1, user descriptor
    g0.y = lds_off;                                         // lds_addr (bytes)
    g0.z = (unsigned int)ga;                                // global_addr[31:0]
    g0.w = (unsigned int)((ga >> 32) & 0x01FFFFFFu) | 0x80000000u; // addr[56:32] | type=2
    v8i g1;
    g1[0] = 0x00020000;                                     // data_size=2 -> 4-byte units
    g1[1] = (int)((tensor_w_dw & 0xFFFFu) << 16);           // tensor_dim0[15:0]
    g1[2] = (int)(((tensor_w_dw >> 16) & 0xFFFFu) | ((tensor_h & 0xFFFFu) << 16));
    g1[3] = (int)(((tensor_h >> 16) & 0xFFFFu) | ((tile_w_dw & 0xFFFFu) << 16)); // tile_dim0
    g1[4] = (int)(tile_h & 0xFFFFu);                        // tile_dim1 (tile_dim2=0)
    g1[5] = (int)stride_dw;                                 // tensor_dim0_stride[31:0]
    g1[6] = 0;                                              // stride[47:32], dim1_stride lo
    g1[7] = 0;
    v4i z4 = {0, 0, 0, 0};
    v8i z8 = {0, 0, 0, 0, 0, 0, 0, 0};
    // clang-23 / amdgpu-toolchain 6-arg form: (g0, g1, g2, g3, g4, cpol)
    __builtin_amdgcn_tensor_load_to_lds(g0, g1, z4, z4, z8, 0);
}

#define LDS_OFF(p) ((unsigned int)(uintptr_t)(p))

// ---------------------------------------------------------------------------
// Prep kernels
// ---------------------------------------------------------------------------
__global__ void cvt_half_kernel(const float* __restrict__ src,
                                _Float16* __restrict__ dst,
                                float scale, int n) {
    int i = blockIdx.x * blockDim.x + threadIdx.x;
    if (i < n) dst[i] = (_Float16)(src[i] * scale);
}

__global__ void logmask_kernel(const float* __restrict__ mask,
                               _Float16* __restrict__ lm, int n) {
    int i = blockIdx.x * blockDim.x + threadIdx.x;
    if (i < n) {
        float v = __logf(mask[i]);
        v = fmaxf(v, -60000.0f);   // exp() of this is 0, matching the FLOAT_MIN clamp
        lm[i] = (_Float16)v;
    }
}

// V[b][k][v] (f32) -> Vt[b][v][k] (f16), tiled transpose through LDS
__global__ __launch_bounds__(256) void transpose_v_kernel(const float* __restrict__ V,
                                                          _Float16* __restrict__ Vt) {
    __shared__ _Float16 tile[64][65];
    int b  = blockIdx.x >> 5;          // LK/64 = 32 tiles per batch
    int k0 = (blockIdx.x & 31) * 64;
    for (int i = threadIdx.x; i < 64 * 64; i += 256) {
        int k = i >> 6, v = i & 63;
        tile[k][v] = (_Float16)V[(size_t)b * LK * DH + (size_t)(k0 + k) * DH + v];
    }
    __syncthreads();
    for (int i = threadIdx.x; i < 64 * 64; i += 256) {
        int v = i >> 6, k = i & 63;
        Vt[(size_t)b * DH * LK + (size_t)v * LK + k0 + k] = tile[k][v];
    }
}

// ---------------------------------------------------------------------------
// Pass 1: per-(b,k) column max M and inverse sum Sinv over the query axis
// grid: B * (LK/128); 256 threads = 8 waves; wave w owns 16 key-columns.
// K tile staged once via TDM; Q tiles double-buffered via TDM.
// ---------------------------------------------------------------------------
__global__ __launch_bounds__(256) void pass1_stats_kernel(
    const _Float16* __restrict__ Qh, const _Float16* __restrict__ Kh,
    const _Float16* __restrict__ LM,
    float* __restrict__ Mg, float* __restrict__ Sg) {
    __shared__ __align__(16) _Float16 kt[128 * DH];   // 16 KB
    __shared__ __align__(16) _Float16 qt[2][16 * DH]; // 2 x 2 KB, double buffered

    const int b   = blockIdx.x >> 4;        // LK/128 = 16 blocks per batch
    const int kb  = (blockIdx.x & 15) * 128;
    const int tid = threadIdx.x;
    const int lane = tid & 31, wv = tid >> 5;
    const int lh = lane >> 4, ll = lane & 15;
    const int myk = kb + 16 * wv;           // this wave's key-column base (global)

    const _Float16* qbase = Qh + (size_t)b * LQ * DH;
    if (wv == 0) {
        // K tile: 128x64 halves contiguous = 4096 dwords, 1-D
        tdm_load_2d(LDS_OFF(kt), Kh + (size_t)b * LK * DH + (size_t)kb * DH,
                    4096, 1, 4096, 4096, 1);
        // first Q tile: 16x64 halves = 512 dwords
        tdm_load_2d(LDS_OFF(qt[0]), qbase, 512, 1, 512, 512, 1);
        __builtin_amdgcn_s_wait_tensorcnt(0);
    }
    __syncthreads();

    float rm = -3.0e38f, rs = 0.0f;

    for (int q0 = 0; q0 < LQ; q0 += 16) {
        const int cur = (q0 >> 4) & 1;
        if (wv == 0 && q0 + 16 < LQ)   // prefetch next Q tile into the other buffer
            tdm_load_2d(LDS_OFF(qt[cur ^ 1]), qbase + (size_t)(q0 + 16) * DH,
                        512, 1, 512, 512, 1);

        v8f acc = {};
        #pragma unroll
        for (int d0 = 0; d0 < DH; d0 += 32) {
            v16h a, bm;
            const _Float16* ap = qt[cur] + ll * DH + d0 + 8 * lh;
            *((v8h*)&a)     = *(const v8h*)ap;
            *((v8h*)&a + 1) = *(const v8h*)(ap + 16);
            const _Float16* bp = kt + (16 * wv + ll) * DH + d0 + 8 * lh;
            *((v8h*)&bm)     = *(const v8h*)bp;
            *((v8h*)&bm + 1) = *(const v8h*)(bp + 16);
            acc = __builtin_amdgcn_wmma_f32_16x16x32_f16(
                false, a, false, bm, (short)0, acc, false, false);
        }

        // D layout: lane holds column k = myk+ll, rows q = q0 + r + 8*lh
        const _Float16* lmp = LM + (size_t)(q0 + 8 * lh) * LK + myk + ll;
        float t[8], tm = -3.0e38f;
        #pragma unroll
        for (int r = 0; r < 8; ++r) {
            t[r] = acc[r] + (float)lmp[(size_t)r * LK];
            tm = fmaxf(tm, t[r]);
        }
        float nm = fmaxf(rm, tm);
        float s = 0.0f;
        #pragma unroll
        for (int r = 0; r < 8; ++r) s += __expf(t[r] - nm);
        rs = rs * __expf(rm - nm) + s;
        rm = nm;

        if (wv == 0 && q0 + 16 < LQ) __builtin_amdgcn_s_wait_tensorcnt(0);
        __syncthreads();   // next buffer ready; all waves done with current buffer
    }

    // combine lane l with l+16 (both hold the same column, different q rows)
    float rm2 = __shfl_xor(rm, 16, 32);
    float rs2 = __shfl_xor(rs, 16, 32);
    float nm  = fmaxf(rm, rm2);
    float tot = rs * __expf(rm - nm) + rs2 * __expf(rm2 - nm);
    if (lane < 16) {
        Mg[(size_t)b * LK + myk + lane] = nm;
        Sg[(size_t)b * LK + myk + lane] = 1.0f / tot;
    }
}

// ---------------------------------------------------------------------------
// Pass 2: out[b,q,:] = sum_k exp(s - M[k]) * Sinv[k] * V[k,:]
// grid: B * (LQ/128); 256 threads = 8 waves; wave w owns 16 query rows.
// Q block via TDM once; K/V strips double-buffered via TDM.
// ---------------------------------------------------------------------------
__global__ __launch_bounds__(256) void pass2_out_kernel(
    const _Float16* __restrict__ Qh, const _Float16* __restrict__ Kh,
    const _Float16* __restrict__ Vt, const _Float16* __restrict__ LM,
    const float* __restrict__ Mg, const float* __restrict__ Sg,
    float* __restrict__ Out) {
    __shared__ __align__(16) _Float16 qt[128 * DH];     // 16 KB
    __shared__ __align__(16) _Float16 kt[2][32 * DH];   // 2 x 4 KB
    __shared__ __align__(16) _Float16 vt[2][DH * 32];   // 2 x 4 KB, [v][k]
    __shared__ __align__(16) _Float16 pt[8][16 * 32];   // 8 KB, wave-private P tiles

    const int b   = blockIdx.x >> 4;        // LQ/128 = 16 blocks per batch
    const int qb  = (blockIdx.x & 15) * 128;
    const int tid = threadIdx.x;
    const int lane = tid & 31, wv = tid >> 5;
    const int lh = lane >> 4, ll = lane & 15;
    const int qrow = 16 * wv;

    const _Float16* kbase  = Kh + (size_t)b * LK * DH;
    const _Float16* vtbase = Vt + (size_t)b * DH * LK;

    if (wv == 0) {
        // Q block: 128x64 halves = 4096 dwords, 1-D
        tdm_load_2d(LDS_OFF(qt), Qh + (size_t)b * LQ * DH + (size_t)qb * DH,
                    4096, 1, 4096, 4096, 1);
        // first K strip: 32x64 halves = 1024 dwords, 1-D
        tdm_load_2d(LDS_OFF(kt[0]), kbase, 1024, 1, 1024, 1024, 1);
        // first V strip: 2-D tile, 8 dwords wide x 64 rows, row stride LK/2 dwords
        tdm_load_2d(LDS_OFF(vt[0]), vtbase, 8, 64, LK / 2, LK / 2, 64);
        __builtin_amdgcn_s_wait_tensorcnt(0);
    }
    __syncthreads();

    v8f o0 = {}, o1 = {}, o2 = {}, o3 = {};

    for (int k0 = 0; k0 < LK; k0 += 32) {
        const int cur = (k0 >> 5) & 1;
        if (wv == 0 && k0 + 32 < LK) {   // issue next strips into the other buffers
            tdm_load_2d(LDS_OFF(kt[cur ^ 1]), kbase + (size_t)(k0 + 32) * DH,
                        1024, 1, 1024, 1024, 1);
            tdm_load_2d(LDS_OFF(vt[cur ^ 1]), vtbase + (k0 + 32),
                        8, 64, LK / 2, LK / 2, 64);
        }
        if (k0 + 64 < LK)   // warm L2 two strips ahead
            __builtin_prefetch(kbase + (size_t)(k0 + 64) * DH + tid * 8, 0, 3);

        // S tiles: 16 q-rows x 32 keys (two 16x16 tiles)
        v8f s0 = {}, s1 = {};
        #pragma unroll
        for (int d0 = 0; d0 < DH; d0 += 32) {
            v16h a, b0, b1;
            const _Float16* ap = qt + (qrow + ll) * DH + d0 + 8 * lh;
            *((v8h*)&a)     = *(const v8h*)ap;
            *((v8h*)&a + 1) = *(const v8h*)(ap + 16);
            const _Float16* bp0 = kt[cur] + ll * DH + d0 + 8 * lh;
            *((v8h*)&b0)     = *(const v8h*)bp0;
            *((v8h*)&b0 + 1) = *(const v8h*)(bp0 + 16);
            const _Float16* bp1 = kt[cur] + (16 + ll) * DH + d0 + 8 * lh;
            *((v8h*)&b1)     = *(const v8h*)bp1;
            *((v8h*)&b1 + 1) = *(const v8h*)(bp1 + 16);
            s0 = __builtin_amdgcn_wmma_f32_16x16x32_f16(false, a, false, b0, (short)0, s0, false, false);
            s1 = __builtin_amdgcn_wmma_f32_16x16x32_f16(false, a, false, b1, (short)0, s1, false, false);
        }

        // P = exp(s + lm - M[k]) * Sinv[k], written to wave-private LDS tile [16 q][32 k]
        const int kc0 = k0 + ll, kc1 = kc0 + 16;
        const float m0 = Mg[(size_t)b * LK + kc0], i0 = Sg[(size_t)b * LK + kc0];
        const float m1 = Mg[(size_t)b * LK + kc1], i1 = Sg[(size_t)b * LK + kc1];
        const _Float16* lmp = LM + (size_t)(qb + qrow + 8 * lh) * LK + k0 + ll;
        _Float16* pw = pt[wv];
        #pragma unroll
        for (int r = 0; r < 8; ++r) {
            float l0 = (float)lmp[(size_t)r * LK];
            float l1 = (float)lmp[(size_t)r * LK + 16];
            float p0 = __expf(s0[r] + l0 - m0) * i0;
            float p1 = __expf(s1[r] + l1 - m1) * i1;
            int m = r + 8 * lh;
            pw[m * 32 + ll]      = (_Float16)p0;
            pw[m * 32 + 16 + ll] = (_Float16)p1;
        }
        asm volatile("s_wait_dscnt 0" ::: "memory");  // wave-private RAW through LDS

        // Re-stripe P as A-fragment (16x32), multiply by V strip (32 x 64)
        v16h pa;
        const _Float16* pap = pw + ll * 32 + 8 * lh;
        *((v8h*)&pa)     = *(const v8h*)pap;
        *((v8h*)&pa + 1) = *(const v8h*)(pap + 16);
        #pragma unroll
        for (int t = 0; t < 4; ++t) {
            v16h bv;
            const _Float16* bp = vt[cur] + (16 * t + ll) * 32 + 8 * lh;
            *((v8h*)&bv)     = *(const v8h*)bp;
            *((v8h*)&bv + 1) = *(const v8h*)(bp + 16);
            v8f* oc = (t == 0) ? &o0 : (t == 1) ? &o1 : (t == 2) ? &o2 : &o3;
            *oc = __builtin_amdgcn_wmma_f32_16x16x32_f16(false, pa, false, bv, (short)0, *oc, false, false);
        }

        if (wv == 0 && k0 + 32 < LK) __builtin_amdgcn_s_wait_tensorcnt(0);
        __syncthreads();   // next buffers ready; everyone done with current buffers
    }

    // Write out tile: 16 q-rows x 64 v-cols
    float* op = Out + (size_t)b * LQ * DH + (size_t)(qb + qrow) * DH;
    #pragma unroll
    for (int r = 0; r < 8; ++r) {
        int m = r + 8 * lh;
        op[(size_t)m * DH + 0  + ll] = o0[r];
        op[(size_t)m * DH + 16 + ll] = o1[r];
        op[(size_t)m * DH + 32 + ll] = o2[r];
        op[(size_t)m * DH + 48 + ll] = o3[r];
    }
}

// ---------------------------------------------------------------------------
extern "C" void kernel_launch(void* const* d_in, const int* in_sizes, int n_in,
                              void* d_out, int out_size, void* d_ws, size_t ws_size,
                              hipStream_t stream) {
    const float* Q    = (const float*)d_in[0];
    const float* K    = (const float*)d_in[1];
    const float* V    = (const float*)d_in[2];
    const float* mask = (const float*)d_in[3];
    float* Out        = (float*)d_out;

    char* ws = (char*)d_ws;
    const size_t qkv_h = (size_t)B_DIM * LQ * DH * sizeof(_Float16);  // 4 MB each
    _Float16* Qh = (_Float16*)(ws);
    _Float16* Kh = (_Float16*)(ws + qkv_h);
    _Float16* Vt = (_Float16*)(ws + 2 * qkv_h);
    _Float16* LM = (_Float16*)(ws + 3 * qkv_h);                       // 8 MB
    float*    Mg = (float*)(ws + 3 * qkv_h + (size_t)LQ * LK * sizeof(_Float16));
    float*    Sg = Mg + (size_t)B_DIM * LK;

    const int nqk = B_DIM * LQ * DH;      // 2,097,152
    const int nlm = LQ * LK;              // 4,194,304

    cvt_half_kernel<<<(nqk + 255) / 256, 256, 0, stream>>>(Q, Qh, 0.125f, nqk); // fold 1/sqrt(64)
    cvt_half_kernel<<<(nqk + 255) / 256, 256, 0, stream>>>(K, Kh, 1.0f, nqk);
    transpose_v_kernel<<<B_DIM * (LK / 64), 256, 0, stream>>>(V, Vt);
    logmask_kernel<<<(nlm + 255) / 256, 256, 0, stream>>>(mask, LM, nlm);

    pass1_stats_kernel<<<B_DIM * (LK / 128), 256, 0, stream>>>(Qh, Kh, LM, Mg, Sg);
    pass2_out_kernel<<<B_DIM * (LQ / 128), 256, 0, stream>>>(Qh, Kh, Vt, LM, Mg, Sg, Out);
}
